// GraphSageLayer_85529978732852
// MI455X (gfx1250) — compile-verified, matrix operand order checked by reading, not run on hardware
//
#include <hip/hip_runtime.h>

typedef _Float16 half16 __attribute__((ext_vector_type(16)));
typedef float    floatx8 __attribute__((ext_vector_type(8)));
typedef int      intx16 __attribute__((ext_vector_type(16)));
typedef unsigned int uintx8 __attribute__((ext_vector_type(8)));

#define N_NODES 16384
#define F_IN 256
#define F_OUT 256

// workspace layout (bytes), all 64B-aligned
static constexpr size_t XH_OFF  = 0;                                  // f16 [16384][256]  (row-major x)
static constexpr size_t XHT_OFF = XH_OFF  + (size_t)N_NODES*F_IN*2;   // f16 [256][16384]  (x transposed)
static constexpr size_t WHT_OFF = XHT_OFF + (size_t)F_IN*N_NODES*2;   // f16 [256][512]    (weight transposed)
static constexpr size_t C1_OFF  = WHT_OFF + (size_t)F_OUT*2*F_IN*2;   // f16 [16384][256]  (neighbor mean)

// ---- CDNA5 async global->LDS copy (ASYNCcnt-tracked), ISA §15.18 op 98 ----
__device__ __forceinline__ void async_copy_b128(unsigned lds_addr, const void* gaddr) {
    asm volatile("global_load_async_to_lds_b128 %0, %1, off"
                 :: "v"(lds_addr), "v"(gaddr)
                 : "memory");
}

#if __has_builtin(__builtin_amdgcn_s_wait_asynccnt)
#define WAIT_ASYNC(n) __builtin_amdgcn_s_wait_asynccnt(n)
#else
#define WAIT_ASYNC(n) asm volatile("s_wait_asynccnt %0" :: "i"(n) : "memory")
#endif

// split workgroup barrier without compiler-forced counter drains
__device__ __forceinline__ void block_barrier() {
#if __has_builtin(__builtin_amdgcn_s_barrier_signal) && __has_builtin(__builtin_amdgcn_s_barrier_wait)
    asm volatile("" ::: "memory");
    __builtin_amdgcn_s_barrier_signal(-1);
    __builtin_amdgcn_s_barrier_wait(-1);
    asm volatile("" ::: "memory");
#else
    __syncthreads();
#endif
}

// ---------------- prep kernels ----------------

__global__ void sage_cvt_x(const float* __restrict__ x, _Float16* __restrict__ xh) {
    int i = blockIdx.x * blockDim.x + threadIdx.x;
    xh[i] = (_Float16)x[i];
}

// 32x32 LDS tile transpose: x[N][256] f32 -> xt[256][N] f16
__global__ void sage_transpose_x(const float* __restrict__ x, _Float16* __restrict__ xt) {
    __shared__ float tile[32][33];
    const int c0 = blockIdx.x * 32;
    const int m0 = blockIdx.y * 32;
    const int tx = threadIdx.x;       // 0..31
    const int ty = threadIdx.y;       // 0..7
#pragma unroll
    for (int i = 0; i < 4; ++i)
        tile[ty * 4 + i][tx] = x[(size_t)(m0 + ty * 4 + i) * F_IN + c0 + tx];
    __syncthreads();
#pragma unroll
    for (int i = 0; i < 4; ++i)
        xt[(size_t)(c0 + ty * 4 + i) * N_NODES + m0 + tx] = (_Float16)tile[tx][ty * 4 + i];
}

// weight [512][256] f32 -> wht [256][512] f16  (wht[n][k] = w[k][n])
__global__ void sage_cvt_w(const float* __restrict__ w, _Float16* __restrict__ wht) {
    int i = blockIdx.x * blockDim.x + threadIdx.x;   // i = n*512 + k
    int n = i >> 9;
    int k = i & 511;
    wht[i] = (_Float16)w[(size_t)k * F_OUT + n];
}

// ---------------- GEMM 1: c1 = (adj @ x) / deg, f16 out ----------------
// block: 256 threads = 8 waves; 64 rows x 256 cols per block, K-step 32.
// 4-buffer async pipeline in LDS (96KB), 3 stages of copies in flight.
// Per substep i: wait ASYNCcnt<=12 (stage i resident), barrier, issue
// stage i+3, compute stage i from LDS (ds_load frags -> WMMA).

__global__ __launch_bounds__(256) void sage_gemm_adj(const int* __restrict__ adj,
                                                     const _Float16* __restrict__ xht,
                                                     _Float16* __restrict__ c1) {
    __shared__ _Float16 Bs[4][256][32];   // [buf][n][k]  16KB per buf
    __shared__ int      As[4][64][32];    // [buf][m][k]   8KB per buf
    __shared__ float    s_deg[64];

    const int lane  = threadIdx.x & 31;
    const int wid   = __builtin_amdgcn_readfirstlane((int)(threadIdx.x >> 5));
    const int wm    = wid >> 1;                 // 0..3
    const int wn    = wid & 1;                  // 0..1
    const int mBlk  = blockIdx.x * 64;
    const int kHalf = (lane >> 4) * 16;         // 0 or 16: K-half this lane holds
    const int nBase = wn * 128 + (lane & 15);   // B-frag column for this lane

    const unsigned ldsB[4] = {
        (unsigned)(size_t)&Bs[0][0][0], (unsigned)(size_t)&Bs[1][0][0],
        (unsigned)(size_t)&Bs[2][0][0], (unsigned)(size_t)&Bs[3][0][0]};
    const unsigned ldsA[4] = {
        (unsigned)(size_t)&As[0][0][0], (unsigned)(size_t)&As[1][0][0],
        (unsigned)(size_t)&As[2][0][0], (unsigned)(size_t)&As[3][0][0]};

    floatx8 acc[8] = {};
    floatx8 accd   = {};

    half16 bOnes;
#pragma unroll
    for (int i = 0; i < 16; ++i) bOnes[i] = (_Float16)1.0f;

    // cooperative async stage of one K-step into buffer `buf`
#define SAGE_STAGE(buf, kk)                                                     \
    do {                                                                        \
        _Pragma("unroll")                                                       \
        for (int j = 0; j < 4; ++j) {            /* B slab: 256 rows x 64B */   \
            const int c  = (wid * 4 + j) * 32 + lane;   /* 0..1023 */           \
            const int n  = c >> 2;                                              \
            const int k4 = c & 3;                                               \
            async_copy_b128(ldsB[buf] + n * 64 + k4 * 16,                       \
                (const char*)xht + ((size_t)n * N_NODES + (kk)) * 2 + k4 * 16); \
        }                                                                       \
        _Pragma("unroll")                                                       \
        for (int j = 0; j < 2; ++j) {            /* A tile: 64 rows x 128B */   \
            const int c  = (wid * 2 + j) * 32 + lane;   /* 0..511 */            \
            const int r  = c >> 3;                                              \
            const int k4 = c & 7;                                               \
            async_copy_b128(ldsA[buf] + r * 128 + k4 * 16,                      \
                (const char*)adj + ((size_t)(mBlk + r) * N_NODES + (kk)) * 4 + k4 * 16); \
        }                                                                       \
    } while (0)

    // consume one staged K-step from LDS buffer `buf`
#define SAGE_COMPUTE(buf)                                                       \
    do {                                                                        \
        const intx16 ai = *(const intx16*)&As[buf][wm * 16 + (lane & 15)][kHalf]; \
        union { uintx8 u; half16 h; } cv;                                       \
        _Pragma("unroll")                                                       \
        for (int i = 0; i < 8; ++i)                                             \
            cv.u[i] = (unsigned)ai[2 * i] * 0x3C00u +                           \
                      (unsigned)ai[2 * i + 1] * 0x3C000000u;                    \
        const half16 af = cv.h;                                                 \
        _Pragma("unroll")                                                       \
        for (int t = 0; t < 8; ++t) {                                           \
            const half16 bf = *(const half16*)&Bs[buf][nBase + t * 16][kHalf];  \
            acc[t] = __builtin_amdgcn_wmma_f32_16x16x32_f16(false, af, false,   \
                         bf, (short)0, acc[t], false, false);                   \
        }                                                                       \
        accd = __builtin_amdgcn_wmma_f32_16x16x32_f16(false, af, false, bOnes,  \
                         (short)0, accd, false, false);                         \
    } while (0)

    // prologue: stages 0..2 (k = 0, 32, 64) in flight
    SAGE_STAGE(0, 0);
    SAGE_STAGE(1, 32);
    SAGE_STAGE(2, 64);

    // steady state: substeps 0..507  (k = 0 .. 16224), all issues in-bounds
    for (int k0 = 0; k0 + 256 <= N_NODES; k0 += 128) {
#define SAGE_SUBSTEP(i)                                                         \
        do {                                                                    \
            WAIT_ASYNC(12);                  /* stage i resident            */  \
            block_barrier();                 /* ... in every wave           */  \
            SAGE_STAGE(((i) + 3) & 3, k0 + (i) * 32 + 96);                      \
            SAGE_COMPUTE((i));                                                  \
        } while (0)
        SAGE_SUBSTEP(0);
        SAGE_SUBSTEP(1);
        SAGE_SUBSTEP(2);
        SAGE_SUBSTEP(3);
#undef SAGE_SUBSTEP
    }

    // tail: substeps 508..511 (k = 16256..16352); waits tighten as issues stop
    WAIT_ASYNC(12); block_barrier(); SAGE_STAGE(3, N_NODES - 32); SAGE_COMPUTE(0);
    WAIT_ASYNC(12); block_barrier(); SAGE_COMPUTE(1);
    WAIT_ASYNC(6);  block_barrier(); SAGE_COMPUTE(2);
    WAIT_ASYNC(0);  block_barrier(); SAGE_COMPUTE(3);
#undef SAGE_STAGE
#undef SAGE_COMPUTE

    // publish deg (column 0 of the ones-accumulator) through LDS
    const int halfOff = (lane >> 4) * 8;        // C layout: VGPR r -> M = r + 8*(lane>=16)
    if (wn == 0 && (lane & 15) == 0) {
#pragma unroll
        for (int r = 0; r < 8; ++r)
            s_deg[wm * 16 + halfOff + r] = accd[r];
    }
    __syncthreads();

    float rdeg[8];
#pragma unroll
    for (int r = 0; r < 8; ++r)
        rdeg[r] = 1.0f / s_deg[wm * 16 + halfOff + r];

#pragma unroll
    for (int t = 0; t < 8; ++t) {
#pragma unroll
        for (int r = 0; r < 8; ++r) {
            const int m = mBlk + wm * 16 + halfOff + r;
            c1[(size_t)m * F_IN + nBase + t * 16] = (_Float16)(acc[t][r] * rdeg[r]);
        }
    }
}

// ---------------- GEMM 2: out = [c1, x] @ weight + bias, f32 out ----------------

__global__ __launch_bounds__(256) void sage_gemm_out(const _Float16* __restrict__ c1h,
                                                     const _Float16* __restrict__ xh,
                                                     const _Float16* __restrict__ wht,
                                                     const float* __restrict__ bias,
                                                     float* __restrict__ out) {
    const int lane  = threadIdx.x & 31;
    const int wid   = __builtin_amdgcn_readfirstlane((int)(threadIdx.x >> 5));
    const int wm    = wid >> 1;
    const int wn    = wid & 1;
    const int rowA  = blockIdx.x * 64 + wm * 16 + (lane & 15);
    const int kHalf = (lane >> 4) * 16;
    const int nBase = wn * 128 + (lane & 15);

    floatx8 acc[8] = {};

    // K = 0..255 -> A = c1 (neighbor mean); K = 256..511 -> A = x.
    for (int k0 = 0; k0 < 512; k0 += 32) {
        const _Float16* aSrc = (k0 < 256) ? (c1h + (size_t)rowA * F_IN + k0)
                                          : (xh  + (size_t)rowA * F_IN + (k0 - 256));
        const half16 af = *(const half16*)(aSrc + kHalf);
        half16 bf[8];
#pragma unroll
        for (int t = 0; t < 8; ++t)
            bf[t] = *(const half16*)(wht + (size_t)(nBase + t * 16) * 512 + k0 + kHalf);
#pragma unroll
        for (int t = 0; t < 8; ++t)
            acc[t] = __builtin_amdgcn_wmma_f32_16x16x32_f16(false, af, false, bf[t],
                                                            (short)0, acc[t], false, false);
    }

    const int halfOff = (lane >> 4) * 8;
#pragma unroll
    for (int t = 0; t < 8; ++t) {
        const float b = bias[nBase + t * 16];
#pragma unroll
        for (int r = 0; r < 8; ++r) {
            const int m = blockIdx.x * 64 + wm * 16 + halfOff + r;
            out[(size_t)m * F_OUT + nBase + t * 16] = acc[t][r] + b;
        }
    }
}

// ---------------- launcher ----------------

extern "C" void kernel_launch(void* const* d_in, const int* in_sizes, int n_in,
                              void* d_out, int out_size, void* d_ws, size_t ws_size,
                              hipStream_t stream) {
    const float* x    = (const float*)d_in[0];
    const int*   adj  = (const int*)d_in[1];
    const float* w    = (const float*)d_in[2];
    const float* bias = (const float*)d_in[3];

    char* ws = (char*)d_ws;
    _Float16* xh  = (_Float16*)(ws + XH_OFF);
    _Float16* xht = (_Float16*)(ws + XHT_OFF);
    _Float16* wht = (_Float16*)(ws + WHT_OFF);
    _Float16* c1h = (_Float16*)(ws + C1_OFF);

    sage_cvt_x<<<(N_NODES * F_IN) / 256, 256, 0, stream>>>(x, xh);
    sage_transpose_x<<<dim3(F_IN / 32, N_NODES / 32), dim3(32, 8), 0, stream>>>(x, xht);
    sage_cvt_w<<<(F_OUT * 2 * F_IN) / 256, 256, 0, stream>>>(w, wht);
    sage_gemm_adj<<<N_NODES / 64, 256, 0, stream>>>(adj, xht, c1h);
    sage_gemm_out<<<N_NODES / 64, 256, 0, stream>>>(c1h, xh, wht, bias, (float*)d_out);
}
